// BiMambaBlock_26139170964328
// MI455X (gfx1250) — matrix operand order repaired and checked by reading, not compiled, not run on hardware
//
#include <hip/hip_runtime.h>
#include <math.h>

// ---- problem constants ----
#define Bsz 4
#define Lsz 2048
#define Dsz 256
#define DIsz 512
#define DSsz 16
#define DRsz 16
#define KKsz 4   // conv taps

typedef __attribute__((ext_vector_type(16))) __bf16 v16bf;
typedef __attribute__((ext_vector_type(8)))  float  v8f;

// ------------------------------------------------------------------
// Elementwise converters
// ------------------------------------------------------------------
__global__ void k_f2bf(const float* __restrict__ src, __bf16* __restrict__ dst, int n) {
    int i = blockIdx.x * blockDim.x + threadIdx.x;
    if (i < n) dst[i] = (__bf16)src[i];
}

// build bf16 x for both directions; dir1 is time-reversed
__global__ void k_build_xbf(const float* __restrict__ x, __bf16* __restrict__ xbf) {
    int i = blockIdx.x * blockDim.x + threadIdx.x;
    const int n = 2 * Bsz * Lsz * Dsz;
    if (i >= n) return;
    int d   = i % Dsz;
    int l   = (i / Dsz) % Lsz;
    int b   = (i / (Dsz * Lsz)) % Bsz;
    int dir = i / (Dsz * Lsz * Bsz);
    int ls  = dir ? (Lsz - 1 - l) : l;
    xbf[i] = (__bf16)x[((size_t)b * Lsz + ls) * Dsz + d];
}

__global__ void k_negexp(const float* __restrict__ a_log, float* __restrict__ aneg, int n) {
    int i = blockIdx.x * blockDim.x + threadIdx.x;
    if (i < n) aneg[i] = -__expf(a_log[i]);
}

// ------------------------------------------------------------------
// Fragment bundle for one k-step of the 32x64 macro tile
// ------------------------------------------------------------------
struct Frags { v16bf a0, a1, b0, b1, b2, b3; };

__device__ __forceinline__ void load_frags(Frags& f,
                                           const __bf16* __restrict__ arow0,
                                           const __bf16* __restrict__ arow1,
                                           const __bf16* __restrict__ brow0,
                                           const __bf16* __restrict__ brow1,
                                           const __bf16* __restrict__ brow2,
                                           const __bf16* __restrict__ brow3,
                                           int k0, int kb, int khalf)
{
    const __bf16* ap0 = arow0 + k0 + kb;
    const __bf16* ap1 = arow1 + k0 + kb;
#pragma unroll
    for (int i = 0; i < 8; ++i) { f.a0[i] = ap0[i]; f.a0[8 + i] = ap0[16 + i]; }
#pragma unroll
    for (int i = 0; i < 8; ++i) { f.a1[i] = ap1[i]; f.a1[8 + i] = ap1[16 + i]; }
    const __bf16* bp0 = brow0 + k0 + kb;
    const __bf16* bp1 = brow1 + k0 + kb;
    const __bf16* bp2 = brow2 + k0 + kb;
    const __bf16* bp3 = brow3 + k0 + kb;
#pragma unroll
    for (int i = 0; i < 16; ++i) f.b0[i] = bp0[(i & 7) + ((i >> 3) << 4)];
#pragma unroll
    for (int i = 0; i < 16; ++i) f.b1[i] = bp1[(i & 7) + ((i >> 3) << 4)];
#pragma unroll
    for (int i = 0; i < 16; ++i) f.b2[i] = bp2[(i & 7) + ((i >> 3) << 4)];
#pragma unroll
    for (int i = 0; i < 16; ++i) f.b3[i] = bp3[(i & 7) + ((i >> 3) << 4)];
}

__device__ __forceinline__ void do_wmma(const Frags& f, v8f acc[2][4])
{
    acc[0][0] = __builtin_amdgcn_wmma_f32_16x16x32_bf16(false, f.a0, false, f.b0, (short)0, acc[0][0], false, false);
    acc[1][0] = __builtin_amdgcn_wmma_f32_16x16x32_bf16(false, f.a1, false, f.b0, (short)0, acc[1][0], false, false);
    acc[0][1] = __builtin_amdgcn_wmma_f32_16x16x32_bf16(false, f.a0, false, f.b1, (short)0, acc[0][1], false, false);
    acc[1][1] = __builtin_amdgcn_wmma_f32_16x16x32_bf16(false, f.a1, false, f.b1, (short)0, acc[1][1], false, false);
    acc[0][2] = __builtin_amdgcn_wmma_f32_16x16x32_bf16(false, f.a0, false, f.b2, (short)0, acc[0][2], false, false);
    acc[1][2] = __builtin_amdgcn_wmma_f32_16x16x32_bf16(false, f.a1, false, f.b2, (short)0, acc[1][2], false, false);
    acc[0][3] = __builtin_amdgcn_wmma_f32_16x16x32_bf16(false, f.a0, false, f.b3, (short)0, acc[0][3], false, false);
    acc[1][3] = __builtin_amdgcn_wmma_f32_16x16x32_bf16(false, f.a1, false, f.b3, (short)0, acc[1][3], false, false);
}

// ------------------------------------------------------------------
// Register-blocked, software-pipelined WMMA GEMM:
//   C[M,N] = A[M,K] * W[N,K]^T   (bf16 in, f32 acc)
// Each wave owns a 32x64 tile (2x4 sub-tiles, 8 WMMAs per k-step).
// Double-buffered fragments: next k-step's loads are issued before the
// current WMMA burst so load latency hides behind the matrix pipe.
// __launch_bounds__(256,1): favor registers over occupancy so the
// 8 accumulators + 2 fragment buffers (~190 VGPRs) never spill.
// mode: 0=store f32, 1=accumulate f32,
//       2=+bias, leakyReLU(0.01) -> bf16, 3=+bias+Madd (residual) -> f32
// Requires M % 32 == 0, N % 16 == 0, K % 32 == 0.
// ------------------------------------------------------------------
__global__ void __launch_bounds__(256, 1)
k_gemm_wmma(const __bf16* __restrict__ A, const __bf16* __restrict__ W,
            float* __restrict__ Cf, __bf16* __restrict__ Cbf,
            const float* __restrict__ bias, const float* __restrict__ Madd,
            int M, int N, int Kd, int mode)
{
    const int lane = threadIdx.x & 31;
    const int wave = threadIdx.x >> 5;
    const int tiles_n = N >> 4;
    const int bt_m = M >> 5;                 // 32-row macro tiles
    const int bt_n = (tiles_n + 3) >> 2;     // 64-col macro tiles
    int tile = blockIdx.x * (blockDim.x >> 5) + wave;
    if (tile >= bt_m * bt_n) return;
    const int tm = tile / bt_n;
    const int tn = tile % bt_n;

    const int half  = lane >> 4;        // 0 or 1
    const int l15   = lane & 15;
    const int kb    = half * 8;         // A-matrix K sub-offset
    const int khalf = half * 16;        // passed but unused; B also uses kb

    const __bf16* arow0 = A + (size_t)(tm * 32 + l15) * Kd;
    const __bf16* arow1 = arow0 + (size_t)16 * Kd;

    bool nvalid[4];
    const __bf16* brow[4];
#pragma unroll
    for (int j = 0; j < 4; ++j) {
        int nt = tn * 4 + j;
        nvalid[j] = nt < tiles_n;
        int ntc = nvalid[j] ? nt : (tiles_n - 1);   // clamp: address stays legal
        brow[j] = W + (size_t)(ntc * 16 + l15) * Kd;
    }

    v8f acc[2][4];
#pragma unroll
    for (int i = 0; i < 2; ++i)
#pragma unroll
        for (int j = 0; j < 4; ++j) acc[i][j] = (v8f){};

    Frags cur, nxt;
    load_frags(cur, arow0, arow1, brow[0], brow[1], brow[2], brow[3], 0, kb, khalf);

    for (int k0 = 32; k0 < Kd; k0 += 32) {
        // issue next k-step's loads first; they are independent of the
        // current fragments, so the WMMA burst below hides their latency
        load_frags(nxt, arow0, arow1, brow[0], brow[1], brow[2], brow[3], k0, kb, khalf);
        do_wmma(cur, acc);
        cur = nxt;
    }
    do_wmma(cur, acc);

#pragma unroll
    for (int i = 0; i < 2; ++i) {
        const int rbase = tm * 32 + i * 16 + 8 * half;
#pragma unroll
        for (int j = 0; j < 4; ++j) {
            if (!nvalid[j]) continue;
            const int col = (tn * 4 + j) * 16 + l15;
#pragma unroll
            for (int r = 0; r < 8; ++r) {
                size_t o = (size_t)(rbase + r) * N + col;
                float v = acc[i][j][r];
                if (mode == 0)      { Cf[o] = v; }
                else if (mode == 1) { Cf[o] += v; }
                else if (mode == 2) { v += bias[col]; v = v > 0.f ? v : 0.01f * v; Cbf[o] = (__bf16)v; }
                else                { v += bias[col] + Madd[o]; Cf[o] = v; }
            }
        }
    }
}

// ------------------------------------------------------------------
// Causal depthwise conv (K=4) + bias + SiLU. Reads xc half of xz.
// ------------------------------------------------------------------
__global__ void k_conv_silu(const float* __restrict__ xz, const float* __restrict__ conv_w,
                            const float* __restrict__ conv_b,
                            float* __restrict__ xc, __bf16* __restrict__ xcbf)
{
    int i = blockIdx.x * blockDim.x + threadIdx.x;
    const int n = 2 * Bsz * Lsz * DIsz;
    if (i >= n) return;
    int d   = i % DIsz;
    int l   = (i / DIsz) % Lsz;
    int b   = (i / (DIsz * Lsz)) % Bsz;
    int dir = i / (DIsz * Lsz * Bsz);
    size_t rowbase = ((size_t)(dir * Bsz + b) * Lsz);
    float acc = conv_b[dir * DIsz + d];
#pragma unroll
    for (int k = 0; k < KKsz; ++k) {
        int ls = l - (KKsz - 1) + k;
        if (ls >= 0)
            acc += xz[(rowbase + ls) * (2 * DIsz) + d] * conv_w[(dir * DIsz + d) * KKsz + k];
    }
    float s = acc / (1.f + __expf(-acc));   // silu
    xc[i] = s;
    xcbf[i] = (__bf16)s;
}

// ------------------------------------------------------------------
// dt = softplus(dt_low @ dt_proj_w^T + dt_proj_b)   (K=16, scalar)
// ------------------------------------------------------------------
__global__ void k_dt(const float* __restrict__ xdbl, const float* __restrict__ dtw,
                     const float* __restrict__ dtb, float* __restrict__ dtf)
{
    int i = blockIdx.x * blockDim.x + threadIdx.x;
    const int n = 2 * Bsz * Lsz * DIsz;
    if (i >= n) return;
    int d   = i % DIsz;
    int l   = (i / DIsz) % Lsz;
    int b   = (i / (DIsz * Lsz)) % Bsz;
    int dir = i / (DIsz * Lsz * Bsz);
    size_t row = ((size_t)(dir * Bsz + b) * Lsz + l);
    float acc = dtb[dir * DIsz + d];
    const float* xr = xdbl + row * (DRsz + 2 * DSsz);
    const float* wr = dtw + (size_t)(dir * DIsz + d) * DRsz;
#pragma unroll
    for (int r = 0; r < DRsz; ++r) acc += xr[r] * wr[r];
    dtf[i] = acc > 20.f ? acc : log1pf(__expf(acc));
}

// ------------------------------------------------------------------
// Selective scan: one thread per (dir,b,d), h[16] in registers,
// B/C vectors staged in LDS per timestep. Fuses +xc*Dp and *silu(z),
// emits bf16 for the out_proj WMMA GEMM.
// block = 128 channels; grid = 2*Bsz*(DIsz/128) = 32
// ------------------------------------------------------------------
__global__ void k_scan(const float* __restrict__ dtf, const float* __restrict__ xc,
                       const float* __restrict__ xz, const float* __restrict__ xdbl,
                       const float* __restrict__ aneg, const float* __restrict__ Dp,
                       __bf16* __restrict__ ybf)
{
    int blk   = blockIdx.x;
    int dpart = blk & 3;
    int b     = (blk >> 2) & 3;
    int dir   = blk >> 4;
    int d     = dpart * 128 + threadIdx.x;

    __shared__ float sBC[2 * DSsz];

    float as[DSsz], h[DSsz];
#pragma unroll
    for (int s = 0; s < DSsz; ++s) {
        as[s] = aneg[(size_t)(dir * DIsz + d) * DSsz + s];
        h[s] = 0.f;
    }
    const float dpv = Dp[dir * DIsz + d];
    const size_t base = (size_t)(dir * Bsz + b) * Lsz;

    for (int t = 0; t < Lsz; ++t) {
        size_t rb = base + t;
        if (threadIdx.x < 2 * DSsz)
            sBC[threadIdx.x] = xdbl[rb * (DRsz + 2 * DSsz) + DRsz + threadIdx.x];
        __syncthreads();

        float dt  = dtf[rb * DIsz + d];
        float xcv = xc[rb * DIsz + d];
        float y = 0.f;
#pragma unroll
        for (int s = 0; s < DSsz; ++s) {
            float dA = __expf(dt * as[s]);
            h[s] = dA * h[s] + dt * sBC[s] * xcv;
            y += h[s] * sBC[DSsz + s];
        }
        y += xcv * dpv;
        float z  = xz[rb * (2 * DIsz) + DIsz + d];
        y *= z / (1.f + __expf(-z));        // * silu(z)
        ybf[rb * DIsz + d] = (__bf16)y;
        __syncthreads();
    }
}

// ------------------------------------------------------------------
// LayerNorm over D=256: one block per (b,l) row -> bf16 for MLP
// ------------------------------------------------------------------
__global__ void k_ln(const float* __restrict__ m, const float* __restrict__ g,
                     const float* __restrict__ be, __bf16* __restrict__ hbf)
{
    __shared__ float red[Dsz];
    int row = blockIdx.x, tid = threadIdx.x;
    float v = m[(size_t)row * Dsz + tid];
    red[tid] = v; __syncthreads();
    for (int s = Dsz / 2; s > 0; s >>= 1) { if (tid < s) red[tid] += red[tid + s]; __syncthreads(); }
    float mu = red[0] * (1.f / Dsz); __syncthreads();
    float dv = v - mu;
    red[tid] = dv * dv; __syncthreads();
    for (int s = Dsz / 2; s > 0; s >>= 1) { if (tid < s) red[tid] += red[tid + s]; __syncthreads(); }
    float var = red[0] * (1.f / Dsz);
    float hn = dv * rsqrtf(var + 1e-5f) * g[tid] + be[tid];
    hbf[(size_t)row * Dsz + tid] = (__bf16)hn;
}

// ------------------------------------------------------------------
static inline int cdiv(int a, int b) { return (a + b - 1) / b; }

extern "C" void kernel_launch(void* const* d_in, const int* in_sizes, int n_in,
                              void* d_out, int out_size, void* d_ws, size_t ws_size,
                              hipStream_t stream)
{
    (void)in_sizes; (void)n_in; (void)out_size; (void)ws_size;
    const float* x    = (const float*)d_in[0];
    const float* wip  = (const float*)d_in[1];
    const float* cw   = (const float*)d_in[2];
    const float* cb   = (const float*)d_in[3];
    const float* wxp  = (const float*)d_in[4];
    const float* dtw  = (const float*)d_in[5];
    const float* dtb  = (const float*)d_in[6];
    const float* alog = (const float*)d_in[7];
    const float* Dp   = (const float*)d_in[8];
    const float* wop  = (const float*)d_in[9];
    const float* lng  = (const float*)d_in[10];
    const float* lnb  = (const float*)d_in[11];
    const float* w1   = (const float*)d_in[12];
    const float* b1   = (const float*)d_in[13];
    const float* w2   = (const float*)d_in[14];
    const float* b2   = (const float*)d_in[15];
    float* out = (float*)d_out;

    const int BLD  = Bsz * Lsz * Dsz;       // 2,097,152
    const int BLDI = Bsz * Lsz * DIsz;      // 4,194,304
    const int M    = Bsz * Lsz;             // 8192 rows

    // ---- carve workspace ----
    size_t off = 0;
    auto carve = [&](size_t bytes) -> char* {
        char* p = (char*)d_ws + off;
        off += (bytes + 255) & ~(size_t)255;
        return p;
    };
    __bf16* xbf    = (__bf16*)carve((size_t)2 * BLD * 2);
    __bf16* wipbf  = (__bf16*)carve((size_t)2 * 2 * DIsz * Dsz * 2);
    __bf16* wxpbf  = (__bf16*)carve((size_t)2 * 48 * DIsz * 2);
    __bf16* wopbf  = (__bf16*)carve((size_t)2 * Dsz * DIsz * 2);
    __bf16* w1bf   = (__bf16*)carve((size_t)2 * Dsz * Dsz * 2);
    __bf16* w2bf   = (__bf16*)carve((size_t)Dsz * 2 * Dsz * 2);
    float*  aneg   = (float*) carve((size_t)2 * DIsz * DSsz * 4);
    float*  xz     = (float*) carve((size_t)2 * M * 2 * DIsz * 4);
    float*  xc     = (float*) carve((size_t)2 * BLDI * 4);
    __bf16* xcbf   = (__bf16*)carve((size_t)2 * BLDI * 2);
    float*  xdbl   = (float*) carve((size_t)2 * M * 48 * 4);
    float*  dtf    = (float*) carve((size_t)2 * BLDI * 4);
    __bf16* ybf    = (__bf16*)carve((size_t)2 * BLDI * 2);
    float*  mbuf   = (float*) carve((size_t)BLD * 4);
    __bf16* hbf    = (__bf16*)carve((size_t)BLD * 2);
    __bf16* p1bf   = (__bf16*)carve((size_t)Bsz * Lsz * 2 * Dsz * 2);

    // tiles for 32x64 macro-tiling, 8 waves (256 thr) per block
    auto gtiles = [](int M_, int N_) {
        int tn = N_ >> 4;
        return (M_ >> 5) * ((tn + 3) >> 2);
    };

    // ---- 0: conversions ----
    k_build_xbf<<<cdiv(2 * BLD, 256), 256, 0, stream>>>(x, xbf);
    k_f2bf<<<cdiv(2 * 2 * DIsz * Dsz, 256), 256, 0, stream>>>(wip, wipbf, 2 * 2 * DIsz * Dsz);
    k_f2bf<<<cdiv(2 * 48 * DIsz, 256), 256, 0, stream>>>(wxp, wxpbf, 2 * 48 * DIsz);
    k_f2bf<<<cdiv(2 * Dsz * DIsz, 256), 256, 0, stream>>>(wop, wopbf, 2 * Dsz * DIsz);
    k_f2bf<<<cdiv(2 * Dsz * Dsz, 256), 256, 0, stream>>>(w1, w1bf, 2 * Dsz * Dsz);
    k_f2bf<<<cdiv(Dsz * 2 * Dsz, 256), 256, 0, stream>>>(w2, w2bf, Dsz * 2 * Dsz);
    k_negexp<<<cdiv(2 * DIsz * DSsz, 256), 256, 0, stream>>>(alog, aneg, 2 * DIsz * DSsz);

    // ---- 1: in_proj GEMM per dir: (M,256) x (1024,256)^T -> xz ----
    for (int dir = 0; dir < 2; ++dir) {
        k_gemm_wmma<<<cdiv(gtiles(M, 2 * DIsz), 8), 256, 0, stream>>>(
            xbf + (size_t)dir * BLD, wipbf + (size_t)dir * 2 * DIsz * Dsz,
            xz + (size_t)dir * M * 2 * DIsz, nullptr, nullptr, nullptr,
            M, 2 * DIsz, Dsz, 0);
    }

    // ---- 2: causal conv + silu ----
    k_conv_silu<<<cdiv(2 * BLDI, 256), 256, 0, stream>>>(xz, cw, cb, xc, xcbf);

    // ---- 3: x_proj GEMM per dir: (M,512) x (48,512)^T -> xdbl ----
    for (int dir = 0; dir < 2; ++dir) {
        k_gemm_wmma<<<cdiv(gtiles(M, 48), 8), 256, 0, stream>>>(
            xcbf + (size_t)dir * BLDI, wxpbf + (size_t)dir * 48 * DIsz,
            xdbl + (size_t)dir * M * 48, nullptr, nullptr, nullptr,
            M, 48, DIsz, 0);
    }

    // ---- 4: dt projection + softplus ----
    k_dt<<<cdiv(2 * BLDI, 256), 256, 0, stream>>>(xdbl, dtw, dtb, dtf);

    // ---- 5: selective scan (fused +xc*Dp, *silu(z)) -> ybf ----
    k_scan<<<32, 128, 0, stream>>>(dtf, xc, xz, xdbl, aneg, Dp, ybf);

    // ---- 6: out_proj GEMM: dir0 writes m, dir1 accumulates ----
    for (int dir = 0; dir < 2; ++dir) {
        k_gemm_wmma<<<cdiv(gtiles(M, Dsz), 8), 256, 0, stream>>>(
            ybf + (size_t)dir * BLDI, wopbf + (size_t)dir * Dsz * DIsz,
            mbuf, nullptr, nullptr, nullptr,
            M, Dsz, DIsz, dir == 0 ? 0 : 1);
    }

    // ---- 7: LayerNorm -> hbf ----
    k_ln<<<M, Dsz, 0, stream>>>(mbuf, lng, lnb, hbf);

    // ---- 8: MLP1: (M,256)x(512,256)^T, +b1, leakyReLU -> p1bf ----
    k_gemm_wmma<<<cdiv(gtiles(M, 2 * Dsz), 8), 256, 0, stream>>>(
        hbf, w1bf, nullptr, p1bf, b1, nullptr, M, 2 * Dsz, Dsz, 2);

    // ---- 9: MLP2: (M,512)x(256,512)^T, +b2, +m residual -> out ----
    k_gemm_wmma<<<cdiv(gtiles(M, Dsz), 8), 256, 0, stream>>>(
        p1bf, w2bf, out, nullptr, b2, mbuf, M, Dsz, 2 * Dsz, 3);
}
